// MPNNBlockSingleEdge_35192962023429
// MI455X (gfx1250) — compile-verified, matrix operand order checked by reading, not run on hardware
//
#include <hip/hip_runtime.h>
#include <hip/hip_bf16.h>
#include <stdint.h>

#define NNODES 50000
#define NEDGES 640000
#define DIM 128
#define EDGE_TILES_PER_BLOCK 8   // 8 x 64 edges per block

typedef __attribute__((ext_vector_type(16))) __bf16 v16bf;
typedef __attribute__((ext_vector_type(8)))  float  v8f;

// ---------------------------------------------------------------------------
// B-fragment packing for V_WMMA_F32_16X16X32_BF16 (wave32).
// Element (k, n) of a 128x128 weight (used as B, K-major: out[n] = sum_k x[k]W[k][n])
// lives at fragment (kk = k>>5, nt = n>>4), lane = (n&15) | (((k>>3)&1)<<4),
// j = (k&7) | (((k>>4)&1)<<3)   -- mirrors the documented 16-bit A layout.
// Fragments stored contiguously: 16 bf16 (32B) per lane.
// ---------------------------------------------------------------------------
__device__ __forceinline__ int pack_index(int k, int n) {
  int kk = k >> 5, kb = k & 31;
  int hi = (kb >> 3) & 1;
  int j  = (kb & 7) | (((kb >> 4) & 1) << 3);
  int lane = (n & 15) | (hi << 4);
  int nt = n >> 4;
  return (((kk * 8 + nt) * 32) + lane) * 16 + j;
}

// Build an A fragment (16x32 bf16 slice) for this lane from an fp32 row.
__device__ __forceinline__ v16bf load_a_frag(const float* __restrict__ row, int kk, int hi) {
  const float4* p = (const float4*)(row + kk * 32 + hi * 8);
  float4 x0 = p[0], x1 = p[1], x2 = p[4], x3 = p[5];
  v16bf a;
  a[0]  = (__bf16)x0.x; a[1]  = (__bf16)x0.y; a[2]  = (__bf16)x0.z; a[3]  = (__bf16)x0.w;
  a[4]  = (__bf16)x1.x; a[5]  = (__bf16)x1.y; a[6]  = (__bf16)x1.z; a[7]  = (__bf16)x1.w;
  a[8]  = (__bf16)x2.x; a[9]  = (__bf16)x2.y; a[10] = (__bf16)x2.z; a[11] = (__bf16)x2.w;
  a[12] = (__bf16)x3.x; a[13] = (__bf16)x3.y; a[14] = (__bf16)x3.z; a[15] = (__bf16)x3.w;
  return a;
}

// --------------------------------------------------------------------------
// Async 32KB weight stage: GLOBAL_LOAD_ASYNC_TO_LDS_B128 (ASYNCcnt tracked).
// Each thread issues 16 x 16B async copies; no VGPR round-trip, no ds_store.
// LDS byte offset = low 32 bits of the generic shared-pointer (LDS aperture).
// --------------------------------------------------------------------------
__device__ __forceinline__ void stage_w_async(__bf16* lds, const __bf16* g, int tid) {
  uint32_t l = (uint32_t)(uintptr_t)lds + (uint32_t)tid * 16u;
  unsigned long long gp = (unsigned long long)(uintptr_t)g + (unsigned long long)tid * 16ull;
#pragma unroll
  for (int i = 0; i < 16; ++i) {
    asm volatile("global_load_async_to_lds_b128 %0, %1, off"
                 :: "v"(l + (uint32_t)(i * 2048)),
                    "v"(gp + (unsigned long long)(i * 2048))
                 : "memory");
  }
}
__device__ __forceinline__ void wait_async_all() {
  asm volatile("s_wait_asynccnt 0" ::: "memory");
}

// Native CDNA5 float atomic max (no-return form, STOREcnt-tracked).
__device__ __forceinline__ void atomicMaxF32(float* addr, float val) {
  asm volatile("global_atomic_max_num_f32 %0, %1, off"
               :: "v"((unsigned long long)(uintptr_t)addr), "v"(val)
               : "memory");
}

// ---------------------------------------------------------------------------
// Kernel W: fold weights.  Wc = Wue@Wmu + Wve@Wmv ; bc = bue@Wmu + bve@Wmv + bmu + bmv.
// Pack Wc, Wmu, Wmv, Wm into bf16 WMMA-fragment order. 64 blocks x 256 thr.
// ---------------------------------------------------------------------------
__global__ void pack_weights_kernel(const float* __restrict__ Wue, const float* __restrict__ Wve,
                                    const float* __restrict__ Wmu, const float* __restrict__ Wmv,
                                    const float* __restrict__ Wm,
                                    const float* __restrict__ bue, const float* __restrict__ bve,
                                    const float* __restrict__ bmu, const float* __restrict__ bmv,
                                    __bf16* __restrict__ Wpc, __bf16* __restrict__ Wpmu,
                                    __bf16* __restrict__ Wpmv, __bf16* __restrict__ Wpm,
                                    float* __restrict__ bc) {
  int tid = blockIdx.x * blockDim.x + threadIdx.x;  // 0..16383
  int k = tid >> 7, n = tid & 127;
  float acc = 0.f;
  for (int j = 0; j < DIM; ++j)
    acc += Wue[k * DIM + j] * Wmu[j * DIM + n] + Wve[k * DIM + j] * Wmv[j * DIM + n];
  int pi = pack_index(k, n);
  Wpc[pi]  = (__bf16)acc;
  Wpmu[pi] = (__bf16)Wmu[k * DIM + n];
  Wpmv[pi] = (__bf16)Wmv[k * DIM + n];
  Wpm[pi]  = (__bf16)Wm[k * DIM + n];
  if (k == 0) {
    float b = bmu[n] + bmv[n];
    for (int j = 0; j < DIM; ++j)
      b += bue[j] * Wmu[j * DIM + n] + bve[j] * Wmv[j * DIM + n];
    bc[n] = b;
  }
}

__global__ void init_agg_kernel(float* __restrict__ agg) {
  int i = blockIdx.x * blockDim.x + threadIdx.x;
  if (i < NNODES * DIM) agg[i] = -__builtin_inff();
}

// ---------------------------------------------------------------------------
// Kernel N: hu = h@Wmu, hv = h@Wmv, h1 = h@Wm  (biases deferred to finalize).
// 4 waves/block, 16 rows/wave, 8 N-tiles x 4 K-steps = 32 WMMAs per weight.
// ---------------------------------------------------------------------------
__global__ void __launch_bounds__(128) node_gemm_kernel(
    const float* __restrict__ h, const __bf16* __restrict__ Wp,  // [3][16384] packed
    float* __restrict__ hu, float* __restrict__ hv, float* __restrict__ h1out) {
  __shared__ __align__(32) __bf16 sw[16384];
  int tid = threadIdx.x;
  int wave = tid >> 5, lane = tid & 31;
  int hi = lane >> 4, lm = lane & 15;
  int rowBase = blockIdx.x * 64 + wave * 16;
  int arow = rowBase + lm;

  v16bf afrag[4];
  if (arow < NNODES) {
    const float* rp = h + (size_t)arow * DIM;
#pragma unroll
    for (int kk = 0; kk < 4; ++kk) afrag[kk] = load_a_frag(rp, kk, hi);
  } else {
    v16bf z;
#pragma unroll
    for (int j = 0; j < 16; ++j) z[j] = (__bf16)0.0f;
#pragma unroll
    for (int kk = 0; kk < 4; ++kk) afrag[kk] = z;
  }

  float* outs[3] = {hu, hv, h1out};
  const v8f vzero = {0.f, 0.f, 0.f, 0.f, 0.f, 0.f, 0.f, 0.f};
  for (int w = 0; w < 3; ++w) {
    __syncthreads();                   // LDS consumers of previous weight done
    stage_w_async(sw, Wp + w * 16384, tid);
    wait_async_all();
    __syncthreads();                   // all waves' async stages visible
    const v16bf* bf = (const v16bf*)sw;
    v8f acc[8];
#pragma unroll
    for (int nt = 0; nt < 8; ++nt) acc[nt] = vzero;
#pragma unroll
    for (int kk = 0; kk < 4; ++kk) {
#pragma unroll
      for (int nt = 0; nt < 8; ++nt) {
        v16bf b = bf[(kk * 8 + nt) * 32 + lane];
        acc[nt] = __builtin_amdgcn_wmma_f32_16x16x32_bf16(
            false, afrag[kk], false, b, (short)0, acc[nt], false, false);
      }
    }
    float* o = outs[w];
#pragma unroll
    for (int r = 0; r < 8; ++r) {
      int row = rowBase + hi * 8 + r;
      if (row < NNODES) {
#pragma unroll
        for (int nt = 0; nt < 8; ++nt)
          o[(size_t)row * DIM + nt * 16 + lm] = acc[nt][r];
      }
    }
  }
}

// ---------------------------------------------------------------------------
// Kernel E: each block owns 8 x 64-edge tiles; W_c staged to LDS ONCE (async),
// then a software-pipelined loop: while tile t runs its 32 WMMAs + atomic-max
// epilogue, tile t+1's indices / hu prefetches / A-fragments are in flight.
// msg contribution per lane: acc + hu[src]; scatter via global_atomic_max_num_f32
// (fp max is associative -> deterministic despite atomic ordering).
// ---------------------------------------------------------------------------
__device__ __forceinline__ void load_edge_tile(
    const float* __restrict__ e, const int* __restrict__ src, const int* __restrict__ dst,
    const float* __restrict__ hu, int rowBase, int lm, int hi,
    v16bf afrag[4], int sIdx[8], int dIdx[8]) {
#pragma unroll
  for (int r = 0; r < 8; ++r) {
    int edge = rowBase + hi * 8 + r;
    sIdx[r] = src[edge];
    dIdx[r] = dst[edge];
  }
#pragma unroll
  for (int r = 0; r < 8; ++r)
    __builtin_prefetch(hu + (size_t)sIdx[r] * DIM + lm * 8, 0, 1);
  const float* rp = e + (size_t)(rowBase + lm) * DIM;
#pragma unroll
  for (int kk = 0; kk < 4; ++kk) afrag[kk] = load_a_frag(rp, kk, hi);
}

__global__ void __launch_bounds__(128) edge_msg_kernel(
    const float* __restrict__ e, const int* __restrict__ src, const int* __restrict__ dst,
    const __bf16* __restrict__ Wpc, const float* __restrict__ hu, float* __restrict__ agg) {
  __shared__ __align__(32) __bf16 sw[16384];
  int tid = threadIdx.x;
  stage_w_async(sw, Wpc, tid);         // async 32KB weight stage, overlaps below
  int wave = tid >> 5, lane = tid & 31;
  int hi = lane >> 4, lm = lane & 15;
  int tbase = blockIdx.x * (EDGE_TILES_PER_BLOCK * 64) + wave * 16;

  v16bf afrag[4];
  int sIdx[8], dIdx[8];
  load_edge_tile(e, src, dst, hu, tbase, lm, hi, afrag, sIdx, dIdx);

  wait_async_all();
  __syncthreads();

  const v16bf* bf = (const v16bf*)sw;
  const v8f vzero = {0.f, 0.f, 0.f, 0.f, 0.f, 0.f, 0.f, 0.f};

  for (int t = 0; t < EDGE_TILES_PER_BLOCK; ++t) {
    // Start next tile's loads before doing this tile's compute/epilogue.
    v16bf nfrag[4];
    int nS[8], nD[8];
    if (t < EDGE_TILES_PER_BLOCK - 1)
      load_edge_tile(e, src, dst, hu, tbase + (t + 1) * 64, lm, hi, nfrag, nS, nD);

    v8f acc[8];
#pragma unroll
    for (int nt = 0; nt < 8; ++nt) acc[nt] = vzero;
#pragma unroll
    for (int kk = 0; kk < 4; ++kk) {
#pragma unroll
      for (int nt = 0; nt < 8; ++nt) {
        v16bf b = bf[(kk * 8 + nt) * 32 + lane];
        acc[nt] = __builtin_amdgcn_wmma_f32_16x16x32_bf16(
            false, afrag[kk], false, b, (short)0, acc[nt], false, false);
      }
    }

#pragma unroll
    for (int r = 0; r < 8; ++r) {
      const float* hus = hu + (size_t)sIdx[r] * DIM + lm;
      float* aggd = agg + (size_t)dIdx[r] * DIM + lm;
#pragma unroll
      for (int nt = 0; nt < 8; ++nt) {
        float val = acc[nt][r] + hus[nt * 16];
        atomicMaxF32(aggd + nt * 16, val);
      }
    }

    if (t < EDGE_TILES_PER_BLOCK - 1) {
#pragma unroll
      for (int kk = 0; kk < 4; ++kk) afrag[kk] = nfrag[kk];
#pragma unroll
      for (int r = 0; r < 8; ++r) { sIdx[r] = nS[r]; dIdx[r] = nD[r]; }
    }
  }
}

// out(=h1) += bm ; if node got messages add agg + hv + bc ; ReLU.
__global__ void finalize_kernel(const float* __restrict__ agg, const float* __restrict__ hv,
                                const float* __restrict__ bc, const float* __restrict__ bm,
                                float* __restrict__ out) {
  int i = blockIdx.x * blockDim.x + threadIdx.x;
  if (i >= NNODES * DIM) return;
  int n = i & (DIM - 1);
  float t = out[i] + bm[n];
  float a = agg[i];
  if (a > -1e30f) t += a + hv[i] + bc[n];
  out[i] = fmaxf(t, 0.0f);
}

extern "C" void kernel_launch(void* const* d_in, const int* in_sizes, int n_in,
                              void* d_out, int out_size, void* d_ws, size_t ws_size,
                              hipStream_t stream) {
  (void)in_sizes; (void)n_in; (void)out_size; (void)ws_size;
  const float* h   = (const float*)d_in[0];
  const float* e   = (const float*)d_in[1];
  const int*   src = (const int*)d_in[2];
  const int*   dst = (const int*)d_in[3];
  const float* Wue = (const float*)d_in[4];
  const float* bue = (const float*)d_in[5];
  const float* Wve = (const float*)d_in[6];
  const float* bve = (const float*)d_in[7];
  const float* Wmu = (const float*)d_in[8];
  const float* bmu = (const float*)d_in[9];
  const float* Wmv = (const float*)d_in[10];
  const float* bmv = (const float*)d_in[11];
  const float* Wm  = (const float*)d_in[12];
  const float* bm  = (const float*)d_in[13];

  char* ws = (char*)d_ws;
  __bf16* Wpc  = (__bf16*)(ws);                  // 32768 B
  __bf16* Wpmu = (__bf16*)(ws + 32768);          // 32768 B
  __bf16* Wpmv = (__bf16*)(ws + 65536);          // 32768 B
  __bf16* Wpm  = (__bf16*)(ws + 98304);          // 32768 B
  float*  bc   = (float*)(ws + 131072);          // 512 B
  float*  hu   = (float*)(ws + 131584);          // 25.6 MB
  float*  hv   = (float*)(ws + 131584 + 25600000);
  float*  agg  = (float*)(ws + 131584 + 51200000);
  float*  out  = (float*)d_out;                  // also holds h1

  pack_weights_kernel<<<64, 256, 0, stream>>>(Wue, Wve, Wmu, Wmv, Wm,
                                              bue, bve, bmu, bmv,
                                              Wpc, Wpmu, Wpmv, Wpm, bc);
  init_agg_kernel<<<(NNODES * DIM + 255) / 256, 256, 0, stream>>>(agg);
  node_gemm_kernel<<<(NNODES + 63) / 64, 128, 0, stream>>>(h, Wpmu, hu, hv, out);
  edge_msg_kernel<<<NEDGES / (64 * EDGE_TILES_PER_BLOCK), 128, 0, stream>>>(
      e, src, dst, Wpc, hu, agg);
  finalize_kernel<<<(NNODES * DIM + 255) / 256, 256, 0, stream>>>(agg, hv, bc, bm, out);
}
